// OptimizedISM_64733747085528
// MI455X (gfx1250) — matrix operand order repaired and checked by reading, not compiled, not run on hardware
//
#include <hip/hip_runtime.h>
#include <stdint.h>

typedef __bf16 v16bf __attribute__((ext_vector_type(16)));
typedef float  v8f   __attribute__((ext_vector_type(8)));

#define QSCALE (1.4426950408889634f / 16.0f)   // log2(e)/T folded into Q

// LDS row strides (bytes), padded +16B to avoid bank conflicts
#define KROW 528
#define VROW 144
#define PROW 144
// double-buffered K and V chunks + per-wave P staging
#define SK0_OFF 0
#define SK1_OFF (64 * KROW)                  // 33792
#define SV0_OFF (2 * 64 * KROW)              // 67584
#define SV1_OFF (SV0_OFF + 256 * VROW)       // 104448
#define SP_OFF  (SV0_OFF + 2 * 256 * VROW)   // 141312
#define SMEM_BYTES (SP_OFF + 4 * 16 * PROW)  // 150528 B  (2 blocks fit in 320KB WGP)

static __device__ __forceinline__ unsigned f2bf(float f) {
  union { float f; unsigned u; } x; x.f = f;
  unsigned r = x.u + 0x7FFFu + ((x.u >> 16) & 1u);   // RNE
  return r >> 16;
}

// Async global->LDS copy, 16B per lane, tracked by ASYNCcnt (CDNA5).
static __device__ __forceinline__ void async_b128(int lds_off, const char* gaddr) {
  asm volatile("global_load_async_to_lds_b128 %0, %1, off"
               :: "v"(lds_off), "v"(gaddr) : "memory");
}
static __device__ __forceinline__ void wait_async0() {
  asm volatile("s_wait_asynccnt 0x0" ::: "memory");
}

// A-fragment (16x32 bf16, rows M=lane%16): lane<16 reads bytes [0,32); lane>=16: [32,64)
static __device__ __forceinline__ v16bf load_A(const char* p, int hi) {
  union { uint4 q[2]; v16bf v; } u;
  u.q[0] = *(const uint4*)(p + 2 * hi);
  u.q[1] = *(const uint4*)(p + 2 * hi + 16);
  return u.v;
}
// B-fragment (32x16 bf16, stored [N][K]): lane<16 reads bytes [0,32); lane>=16: [32,64)
static __device__ __forceinline__ v16bf load_B(const char* p, int hi) {
  union { uint4 q[2]; v16bf v; } u;
  const uint4* s = (const uint4*)(p + 2 * hi);
  u.q[0] = s[0]; u.q[1] = s[1];
  return u.v;
}

// S[b,c,i,j] f32 -> S_jc[b,i,j,c] bf16 (transposed) and S_cj[b,c,i,j] bf16 (cast)
__global__ void prep_S(const float* __restrict__ S,
                       unsigned short* __restrict__ Sjc,
                       unsigned short* __restrict__ Scj) {
  __shared__ unsigned short t[256][66];
  const int b = blockIdx.x >> 6, i = blockIdx.x & 63;
  const int c = threadIdx.x;
  const float4* src = (const float4*)(S + ((size_t)(b * 256 + c) * 64 + i) * 64);
  uint2* outc = (uint2*)(Scj + ((size_t)(b * 256 + c) * 64 + i) * 64);
#pragma unroll
  for (int j4 = 0; j4 < 16; j4++) {
    float4 v = src[j4];
    unsigned b0 = f2bf(v.x), b1 = f2bf(v.y), b2 = f2bf(v.z), b3 = f2bf(v.w);
    uint2 pk; pk.x = b0 | (b1 << 16); pk.y = b2 | (b3 << 16);
    outc[j4] = pk;
    t[c][j4 * 4 + 0] = (unsigned short)b0;
    t[c][j4 * 4 + 1] = (unsigned short)b1;
    t[c][j4 * 4 + 2] = (unsigned short)b2;
    t[c][j4 * 4 + 3] = (unsigned short)b3;
  }
  __syncthreads();
  unsigned short* outj = Sjc + ((size_t)(b * 64 + i) * 64) * 256 + c;
#pragma unroll
  for (int j = 0; j < 64; j++) outj[(size_t)j * 256] = t[c][j];
}

// D[b,c,y,x] f32 -> Q_bf[b,y,x,c] bf16, pre-scaled by log2(e)/16
__global__ void prep_D(const float* __restrict__ D, unsigned short* __restrict__ Qbf) {
  __shared__ unsigned short t[256][34];
  const int b = blockIdx.x >> 5, y = blockIdx.x & 31;
  const int c = threadIdx.x;
  const float4* src = (const float4*)(D + ((size_t)(b * 256 + c) * 32 + y) * 32);
#pragma unroll
  for (int x4 = 0; x4 < 8; x4++) {
    float4 v = src[x4];
    t[c][x4 * 4 + 0] = (unsigned short)f2bf(v.x * QSCALE);
    t[c][x4 * 4 + 1] = (unsigned short)f2bf(v.y * QSCALE);
    t[c][x4 * 4 + 2] = (unsigned short)f2bf(v.z * QSCALE);
    t[c][x4 * 4 + 3] = (unsigned short)f2bf(v.w * QSCALE);
  }
  __syncthreads();
  unsigned short* q = Qbf + ((size_t)(b * 32 + y) * 32) * 256 + c;
#pragma unroll
  for (int x = 0; x < 32; x++) q[(size_t)x * 256] = t[c][x];
}

// One block per (b, h): 4 waves, wave wv owns queries w = wv*16..wv*16+15 of row h.
__global__ __launch_bounds__(128, 1)
void attn_kernel(const float* __restrict__ D, const unsigned short* __restrict__ Qbf,
                 const unsigned short* __restrict__ Sjc, const unsigned short* __restrict__ Scj,
                 float* __restrict__ out) {
  extern __shared__ char smem[];
  const int b = blockIdx.x >> 6, h = blockIdx.x & 63;
  const int tid = threadIdx.x;
  const int lane = tid & 31, wv = tid >> 5;
  const int ln = lane & 15, hi = lane & 16;       // hi in {0,16}
  char* sP = smem + SP_OFF + wv * 16 * PROW;
  const int wq0 = wv * 16;
  const int w = wq0 + ln;

  // --- Q lives entirely in registers: 8 B-fragments (16q x 256c per wave) ---
  v16bf Qf[8];
  {
    const char* qrow = (const char*)Qbf +
        ((size_t)(b * 32 + (h >> 1)) * 32) * 512 + (size_t)(w >> 1) * 512;
#pragma unroll
    for (int kk = 0; kk < 8; kk++) {
      union { uint4 q[2]; v16bf v; } u;
      u.q[0] = *(const uint4*)(qrow + kk * 64 + 2 * hi);
      u.q[1] = *(const uint4*)(qrow + kk * 64 + 2 * hi + 16);
      Qf[kk] = u.v;
    }
  }

  const char* kbase = (const char*)(Sjc + ((size_t)b * 64) * 64 * 256);
  const char* vbase = (const char*)(Scj + ((size_t)b * 256) * 64 * 64);

  // issue async loads of chunk `ic` into buffers at (sk_off, sv_off)
  auto issue_chunk = [&](int ic, int sk_off, int sv_off) {
    const char* ksrc = kbase + (size_t)ic * 32768;   // 64*256 bf16 contiguous
#pragma unroll
    for (int it = 0; it < 16; it++) {
      int idx = tid + it * 128, row = idx >> 5, p = idx & 31;
      async_b128(sk_off + row * KROW + p * 16, ksrc + row * 512 + p * 16);
    }
#pragma unroll
    for (int it = 0; it < 16; it++) {
      int idx = tid + it * 128, row = idx >> 3, p = idx & 7;
      const char* vs = vbase + ((size_t)row * 64 + ic) * 128;  // row c, chunk ic: 64 bf16
      async_b128(sv_off + row * VROW + p * 16, vs + p * 16);
    }
  };

  const v8f vzero = {0.f, 0.f, 0.f, 0.f, 0.f, 0.f, 0.f, 0.f};
  v8f accO[16];
#pragma unroll
  for (int ct = 0; ct < 16; ct++) accO[ct] = vzero;

  // prologue: chunk 0 -> buffer 0
  issue_chunk(0, SK0_OFF, SV0_OFF);

  for (int i = 0; i < 64; i++) {
    const int cur = i & 1;
    char* sK = smem + (cur ? SK1_OFF : SK0_OFF);
    char* sV = smem + (cur ? SV1_OFF : SV0_OFF);

    wait_async0();        // this wave's share of chunk i is in LDS
    __syncthreads();      // everyone's share resident; prior reads of other buffer done
    if (i < 63)           // prefetch chunk i+1 into the other buffer (overlaps compute)
      issue_chunk(i + 1, cur ? SK0_OFF : SK1_OFF, cur ? SV0_OFF : SV1_OFF);

    // GEMM1: P^T[j][q] = K_chunk[j][c] x Q^T[c][q]  (Q pre-scaled by log2e/16)
    v8f accP[4];
#pragma unroll
    for (int jt = 0; jt < 4; jt++) accP[jt] = vzero;
#pragma unroll
    for (int kk = 0; kk < 8; kk++) {
#pragma unroll
      for (int jt = 0; jt < 4; jt++) {
        v16bf Ak = load_A(sK + (jt * 16 + ln) * KROW + kk * 64, hi);
        accP[jt] = __builtin_amdgcn_wmma_f32_16x16x32_bf16(
            false, Ak, false, Qf[kk], (short)0, accP[jt], false, false);
      }
    }

    // Softmax over the 64 j's of this chunk: lane holds 32, partner lane (^16) the rest
    float m = -3.0e38f;
#pragma unroll
    for (int jt = 0; jt < 4; jt++)
#pragma unroll
      for (int r = 0; r < 8; r++) m = fmaxf(m, accP[jt][r]);
    m = fmaxf(m, __shfl_xor(m, 16, 32));
    float s = 0.f;
#pragma unroll
    for (int jt = 0; jt < 4; jt++)
#pragma unroll
      for (int r = 0; r < 8; r++) {
        float e = __builtin_amdgcn_exp2f(accP[jt][r] - m);
        accP[jt][r] = e; s += e;
      }
    s += __shfl_xor(s, 16, 32);
    float inv = __builtin_amdgcn_rcpf(s);

    // Store P as bf16 into per-wave LDS [q][j]; element (m=j,n=q) -> row q byte 2j
    char* prow = sP + ln * PROW + hi;   // hi bytes == half*8 j's * 2B
#pragma unroll
    for (int jt = 0; jt < 4; jt++)
#pragma unroll
      for (int r = 0; r < 8; r += 2) {
        unsigned lo = f2bf(accP[jt][r] * inv);
        unsigned h2 = f2bf(accP[jt][r + 1] * inv);
        *(unsigned*)(prow + jt * 32 + r * 2) = lo | (h2 << 16);
      }

    // GEMM2: O^T[c][q] += V^T[c][j] x P^T[j][q]
#pragma unroll
    for (int jj = 0; jj < 2; jj++) {
      v16bf Bp = load_B(sP + ln * PROW + jj * 64, hi);
#pragma unroll
      for (int ct = 0; ct < 16; ct++) {
        v16bf Av = load_A(sV + (ct * 16 + ln) * VROW + jj * 64, hi);
        accO[ct] = __builtin_amdgcn_wmma_f32_16x16x32_bf16(
            false, Av, false, Bp, (short)0, accO[ct], false, false);
      }
    }
  }

  // Epilogue: Z[b,c,h,w] = O^T + D[b,c,h/2,w/2]
  const float* dres = D + (size_t)b * 256 * 1024 + (h >> 1) * 32 + (w >> 1);
  float* o = out + (size_t)b * 256 * 4096 + h * 64 + w;
  const int mh = hi >> 1;   // 0 or 8
#pragma unroll
  for (int ct = 0; ct < 16; ct++)
#pragma unroll
    for (int r = 0; r < 8; r++) {
      int c = ct * 16 + mh + r;
      o[(size_t)c * 4096] = accO[ct][r] + dres[(size_t)c * 1024];
    }
}

extern "C" void kernel_launch(void* const* d_in, const int* in_sizes, int n_in,
                              void* d_out, int out_size, void* d_ws, size_t ws_size,
                              hipStream_t stream) {
  const float* D = (const float*)d_in[0];   // [4,256,32,32]
  const float* S = (const float*)d_in[1];   // [4,256,64,64]
  float* out = (float*)d_out;               // [4,256,64,64]

  const size_t S_ELEMS = (size_t)4 * 256 * 64 * 64;   // 4M
  unsigned short* Sjc = (unsigned short*)d_ws;        // [b][i][j][c]
  unsigned short* Scj = Sjc + S_ELEMS;                // [b][c][i][j]
  unsigned short* Qbf = Scj + S_ELEMS;                // [b][y][x][c]
  (void)in_sizes; (void)n_in; (void)out_size; (void)ws_size;

  prep_S<<<256, 256, 0, stream>>>(S, Sjc, Scj);
  prep_D<<<128, 256, 0, stream>>>(D, Qbf);
  attn_kernel<<<256, 128, SMEM_BYTES, stream>>>(D, Qbf, Sjc, Scj, out);
}
